// FlaxGemma2Attention_962072674318
// MI455X (gfx1250) — compile-verified, matrix operand order checked by reading, not run on hardware
//
#include <hip/hip_runtime.h>
#include <hip/hip_bf16.h>

// ---------------------------------------------------------------------------
// Gemma2 attention on gfx1250 (MI455X).
// f16 WMMA (v_wmma_f32_16x16x32_f16) for all matmuls; TDM (tensor_load_to_lds)
// stages attention K-tiles into LDS with hardware row padding.
// B=2 S=2048 H=2304 NH=8 NKV=4 HD=256 softcap=50
// ---------------------------------------------------------------------------

typedef __attribute__((ext_vector_type(16))) _Float16 v16h;
typedef __attribute__((ext_vector_type(8)))  _Float16 v8h;
typedef __attribute__((ext_vector_type(4)))  _Float16 v4h;
typedef __attribute__((ext_vector_type(8)))  float    v8f;
typedef __attribute__((ext_vector_type(4)))  float    v4f;
typedef __attribute__((ext_vector_type(4)))  unsigned int u32x4;
typedef __attribute__((ext_vector_type(8)))  int      i32x8;
typedef __attribute__((ext_vector_type(4)))  int      i32x4;

static __device__ __forceinline__ v16h cat8(v8h a, v8h b) {
  return __builtin_shufflevector(a, b, 0,1,2,3,4,5,6,7,8,9,10,11,12,13,14,15);
}

static __device__ __forceinline__ v8f wmma_f16(v16h a, v16h b, v8f c) {
  // D = A(16x32) * B(32x16) + C, fp32 accumulate
  return __builtin_amdgcn_wmma_f32_16x16x32_f16(false, a, false, b, (short)0, c,
                                                false, false);
}

// Hardware tanh (TRANS op). Raw asm bypasses compiler hazard tracking, so we
// append v_nop per ISA 7.4 (1 independent op/V_NOP after TRANS before use).
static __device__ __forceinline__ float fast_tanh(float x) {
#if __has_builtin(__builtin_amdgcn_tanhf)
  return __builtin_amdgcn_tanhf(x);
#else
  float r;
  asm volatile("v_tanh_f32 %0, %1\n\tv_nop" : "=v"(r) : "v"(x));
  return r;
#endif
}

// ---------------------------------------------------------------------------
// GEMM: C[M,N] = A[M,K] * B[K,N], fp32 in/out, f16 WMMA compute.
// WG 256 thr = 8 waves (2 x 4) over a 64x128 tile, BK = 32.
// LDS stride 40 halves (80B): 16B-aligned fragment reads, conflict-free.
// ---------------------------------------------------------------------------
#define GA_STRIDE 40

__global__ __launch_bounds__(256) void gemm_hf_wmma(
    const float* __restrict__ A, const float* __restrict__ B,
    float* __restrict__ C, int M, int N, int K)
{
  __shared__ _Float16 Ash[64  * GA_STRIDE];
  __shared__ _Float16 Bsh[128 * GA_STRIDE];   // stored transposed: [n][k]

  const int tid  = threadIdx.x;
  const int lane = tid & 31;
  const int wave = tid >> 5;
  const int wm   = wave & 1;        // 2 waves over M (32 rows each)
  const int wn   = wave >> 1;       // 4 waves over N (32 cols each)
  const int bm   = blockIdx.y * 64;
  const int bn   = blockIdx.x * 128;
  const int r16  = lane & 15;
  const int lo   = (lane < 16) ? 0 : 8;   // ISA 16-bit A/B fragment K-perm

  v8f acc[2][2] = {};

  for (int k0 = 0; k0 < K; k0 += 32) {
    const bool pf = (k0 + 32) < K;
    // stage A tile 64x32 -> f16 LDS (each thread: 2 x float4)
#pragma unroll
    for (int i = 0; i < 2; ++i) {
      int idx = tid + i * 256;            // float4 index, 512 total
      int r = idx >> 3, c4 = idx & 7;
      const float* src = A + (size_t)(bm + r) * K + k0 + c4 * 4;
      if (pf) __builtin_prefetch(src + 32, 0, 1);
      v4f a = *(const v4f*)src;
      v4h h; h.x = (_Float16)a.x; h.y = (_Float16)a.y;
             h.z = (_Float16)a.z; h.w = (_Float16)a.w;
      *(v4h*)&Ash[r * GA_STRIDE + c4 * 4] = h;
    }
    // stage B tile 32x128 transposed -> f16 LDS (each thread: 4 x float4)
#pragma unroll
    for (int i = 0; i < 4; ++i) {
      int idx = tid + i * 256;            // float4 index, 1024 total
      int r = idx >> 5, c4 = idx & 31;
      const float* src = B + (size_t)(k0 + r) * N + bn + c4 * 4;
      if (pf) __builtin_prefetch(src + (size_t)32 * N, 0, 1);
      v4f bv = *(const v4f*)src;
      int n0 = c4 * 4;
      Bsh[(n0 + 0) * GA_STRIDE + r] = (_Float16)bv.x;
      Bsh[(n0 + 1) * GA_STRIDE + r] = (_Float16)bv.y;
      Bsh[(n0 + 2) * GA_STRIDE + r] = (_Float16)bv.z;
      Bsh[(n0 + 3) * GA_STRIDE + r] = (_Float16)bv.w;
    }
    __syncthreads();

    v16h af[2], bf[2];
#pragma unroll
    for (int mi = 0; mi < 2; ++mi) {
      const _Float16* p = &Ash[(wm * 32 + mi * 16 + r16) * GA_STRIDE + lo];
      af[mi] = cat8(*(const v8h*)p, *(const v8h*)(p + 16));
    }
#pragma unroll
    for (int ni = 0; ni < 2; ++ni) {
      const _Float16* p = &Bsh[(wn * 32 + ni * 16 + r16) * GA_STRIDE + lo];
      bf[ni] = cat8(*(const v8h*)p, *(const v8h*)(p + 16));
    }
#pragma unroll
    for (int mi = 0; mi < 2; ++mi)
#pragma unroll
      for (int ni = 0; ni < 2; ++ni)
        acc[mi][ni] = wmma_f16(af[mi], bf[ni], acc[mi][ni]);
    __syncthreads();
  }

  // writeback: C-layout — VGPR j: M = j (+8 for lanes 16-31), N = lane&15
#pragma unroll
  for (int mi = 0; mi < 2; ++mi) {
#pragma unroll
    for (int ni = 0; ni < 2; ++ni) {
      int r0 = bm + wm * 32 + mi * 16 + ((lane < 16) ? 0 : 8);
      int c  = bn + wn * 32 + ni * 16 + r16;
#pragma unroll
      for (int j = 0; j < 8; ++j)
        C[(size_t)(r0 + j) * N + c] = acc[mi][ni][j];
    }
  }
}

// ---------------------------------------------------------------------------
// RoPE + f16 convert (fuses the 1/sqrt(HD) query scale).
// One thread per (b,s,head,d<128) rotation pair.
// ---------------------------------------------------------------------------
__global__ void rope_f16_kernel(const float* __restrict__ src,
                                const int* __restrict__ pos,
                                _Float16* __restrict__ dst,
                                int heads, float scale, int total)
{
  int idx = blockIdx.x * blockDim.x + threadIdx.x;
  if (idx >= total) return;
  int d  = idx & 127;
  int h  = (idx >> 7) % heads;
  int bs = idx / (heads << 7);
  const float LOG2_10000 = 13.287712379549449f;
  float inv = exp2f(-(float)d * (LOG2_10000 / 128.0f));   // 10000^(-d/128)
  float ang = (float)pos[bs] * inv;
  float sn = __sinf(ang), cs = __cosf(ang);
  size_t base = ((size_t)bs * heads + h) * 256;
  float x0 = src[base + d], x1 = src[base + d + 128];
  dst[base + d]       = (_Float16)((x0 * cs - x1 * sn) * scale);
  dst[base + d + 128] = (_Float16)((x1 * cs + x0 * sn) * scale);
}

__global__ void cvt_f16_kernel(const float* __restrict__ src,
                               _Float16* __restrict__ dst, int total)
{
  int idx = blockIdx.x * blockDim.x + threadIdx.x;
  if (idx < total) dst[idx] = (_Float16)src[idx];
}

// ---------------------------------------------------------------------------
// Flash attention: block = (q-block of 64, head, batch), 128 thr = 4 waves,
// each wave owns 16 query rows. Q frags in registers. K tile is DMA'd into
// LDS by the Tensor Data Mover with hardware row padding (stride 264 halves);
// V^T tile shares the same LDS buffer. Per-wave P scratch does the
// register->A-frag transpose. Softmax matches reference order.
// ---------------------------------------------------------------------------
#define AS  2048
#define AH  8
#define AKV 4
#define AHD 256
#define KST 264   // K tile row stride (halves): 256 + TDM pad of 8 (16B)
#define VST 80    // V^T / P row stride (halves): 160B = 10*16, aligned

__global__ __launch_bounds__(128) void flash_attn_kernel(
    const _Float16* __restrict__ qh, const _Float16* __restrict__ kh,
    const _Float16* __restrict__ vh, const int* __restrict__ amask,
    float* __restrict__ O)
{
  __shared__ _Float16 KV[256 * VST];          // max(64*KST=16896, 256*VST=20480)
  __shared__ _Float16 Psh[4][16 * VST];

  const int tid  = threadIdx.x;
  const int lane = tid & 31;
  const int wave = tid >> 5;                  // 0..3
  const int qb = blockIdx.x, h = blockIdx.y, b = blockIdx.z;
  const int kvh = h >> 1;                     // GQA: 2 query heads per KV head
  const int r16 = lane & 15;
  const int lo  = (lane < 16) ? 0 : 8;
  const int qr0 = qb * 64 + wave * 16;

  // Q fragments (A-matrix), 8 K-steps of 32 covering HD=256; Q pre-scaled.
  v16h qf[8];
  {
    const _Float16* qp = qh + ((size_t)(b * AS + qr0 + r16) * AH + h) * AHD;
#pragma unroll
    for (int ks = 0; ks < 8; ++ks) {
      const _Float16* p = qp + ks * 32 + lo;
      qf[ks] = cat8(*(const v8h*)p, *(const v8h*)(p + 16));
    }
  }

  v8f Oacc[16] = {};
  float m[8], l[8];
#pragma unroll
  for (int j = 0; j < 8; ++j) { m[j] = -1e30f; l[j] = 0.0f; }

  _Float16* Pw = &Psh[wave][0];
  const unsigned int lds_base =
      (unsigned int)(unsigned long long)(const void*)&KV[0];

  for (int kb = 0; kb <= qb; ++kb) {          // causal: skip blocks above diag
    const int kbase = kb * 64;

    // ---- TDM: DMA K tile (64 rows x 256 f16, row stride AKV*AHD) to LDS ----
    // pad_enable: +4 DWORDs (8 halves) every 128 DWORDs (one 512B row) so the
    // LDS image lands at stride KST=264 halves. Wave 0 issues; TDM ignores
    // EXEC and runs per-wave, so one wave + s_wait_tensorcnt + barrier.
    if (wave == 0) {
      unsigned long long ga = (unsigned long long)(const void*)
          (kh + ((size_t)(b * AS + kbase) * AKV + kvh) * AHD);
      u32x4 g0;
      g0.x = 1u;                                  // count=1, user descriptor
      g0.y = lds_base;                            // lds_addr
      g0.z = (unsigned int)ga;                    // global_addr[31:0]
      g0.w = (unsigned int)(ga >> 32) | (2u << 30); // ga[56:32] | type=2
      i32x8 g1;
      g1.s0 = (1 << 16)      // data_size = 1 -> 2 bytes
            | (1 << 20)      // pad_enable
            | (6 << 22)      // pad_interval: every 128 DWORDs (one row)
            | (3 << 25);     // pad_amount: 4 DWORDs = 16B = 8 halves
      g1.s1 = (int)(256u << 16);   // tensor_dim0 = 256 elements (bits 79:48)
      g1.s2 = (int)(2048u << 16);  // tensor_dim1 = 2048 rows (bits 111:80)
      g1.s3 = (int)(256u << 16);   // tile_dim0 = 256 (bits 127:112)
      g1.s4 = 64;                  // tile_dim1 = 64 (bits 143:128)
      g1.s5 = 1024;                // tensor_dim0_stride = AKV*AHD (bits 207:160)
      g1.s6 = 0;
      g1.s7 = 0;
      i32x4 z4 = {0, 0, 0, 0};
      i32x8 z8 = {0, 0, 0, 0, 0, 0, 0, 0};
      // 6-arg form (clang-23 / therock-10.0 headers on this toolchain)
      __builtin_amdgcn_tensor_load_to_lds(g0, g1, z4, z4, z8, 0);
      __builtin_amdgcn_s_wait_tensorcnt(0);
    }
    __syncthreads();

    // ---- S = Q K^T : 4 key tiles x 8 K-steps ----
    v8f sacc[4] = {};
#pragma unroll
    for (int ks = 0; ks < 8; ++ks) {
#pragma unroll
      for (int nt = 0; nt < 4; ++nt) {
        const _Float16* p = &KV[(nt * 16 + r16) * KST + ks * 32 + lo];
        v16h kf = cat8(*(const v8h*)p, *(const v8h*)(p + 16));
        sacc[nt] = wmma_f16(qf[ks], kf, sacc[nt]);
      }
    }

    // ---- mask (big_neg) -> softcap, in reference order ----
#pragma unroll
    for (int nt = 0; nt < 4; ++nt) {
      int key = kbase + nt * 16 + r16;
      int am  = amask[b * AS + key];
#pragma unroll
      for (int j = 0; j < 8; ++j) {
        int qr = qr0 + j + ((lane < 16) ? 0 : 8);
        float sv = fast_tanh(sacc[nt][j] * 0.02f) * 50.0f;
        if (key > qr || am == 0) sv = -50.0f;  // tanh(big_neg/50)*50 == -50
        sacc[nt][j] = sv;
      }
    }

    // ---- online softmax: row stats across 16-lane column groups ----
    float scale[8], newm[8];
#pragma unroll
    for (int j = 0; j < 8; ++j) {
      float rm = fmaxf(fmaxf(sacc[0][j], sacc[1][j]),
                       fmaxf(sacc[2][j], sacc[3][j]));
      rm = fmaxf(rm, __shfl_xor(rm, 1, 32));
      rm = fmaxf(rm, __shfl_xor(rm, 2, 32));
      rm = fmaxf(rm, __shfl_xor(rm, 4, 32));
      rm = fmaxf(rm, __shfl_xor(rm, 8, 32));
      newm[j]  = fmaxf(m[j], rm);
      scale[j] = __expf(m[j] - newm[j]);
      m[j]     = newm[j];
    }
#pragma unroll
    for (int dt = 0; dt < 16; ++dt)
#pragma unroll
      for (int j = 0; j < 8; ++j)
        Oacc[dt][j] *= scale[j];

#pragma unroll
    for (int j = 0; j < 8; ++j) {
      float rs = 0.0f;
      int prow = j + ((lane < 16) ? 0 : 8);
#pragma unroll
      for (int nt = 0; nt < 4; ++nt) {
        float p = __expf(sacc[nt][j] - newm[j]);
        Pw[prow * VST + nt * 16 + r16] = (_Float16)p;  // transpose via LDS
        rs += p;
      }
      rs += __shfl_xor(rs, 1, 32);
      rs += __shfl_xor(rs, 2, 32);
      rs += __shfl_xor(rs, 4, 32);
      rs += __shfl_xor(rs, 8, 32);
      l[j] = l[j] * scale[j] + rs;
    }

    __syncthreads();   // K tile reads done; P in LDS; reuse KV for V^T

    // ---- stage V^T tile (256 dims x 64 keys) ----
#pragma unroll
    for (int i = 0; i < 16; ++i) {
      int idx = tid + i * 128;
      int r = idx >> 5, c8 = idx & 31;
      v8h vv = *(const v8h*)(vh +
          ((size_t)(b * AS + kbase + r) * AKV + kvh) * AHD + c8 * 8);
#pragma unroll
      for (int j = 0; j < 8; ++j)
        KV[(c8 * 8 + j) * VST + r] = vv[j];
    }
    __syncthreads();

    // ---- O += P V : 2 K-steps over 64 keys, 16 dim tiles ----
#pragma unroll
    for (int ks = 0; ks < 2; ++ks) {
      const _Float16* pp = &Pw[r16 * VST + ks * 32 + lo];
      v16h pf = cat8(*(const v8h*)pp, *(const v8h*)(pp + 16));
#pragma unroll
      for (int dt = 0; dt < 16; ++dt) {
        const _Float16* vp = &KV[(dt * 16 + r16) * VST + ks * 32 + lo];
        v16h vfr = cat8(*(const v8h*)vp, *(const v8h*)(vp + 16));
        Oacc[dt] = wmma_f16(pf, vfr, Oacc[dt]);
      }
    }
    __syncthreads();   // before next iteration overwrites KV
  }

  // ---- finalize: divide by softmax denominator, write [B,S,NH,HD] fp32 ----
#pragma unroll
  for (int j = 0; j < 8; ++j) l[j] = 1.0f / l[j];
#pragma unroll
  for (int dt = 0; dt < 16; ++dt) {
#pragma unroll
    for (int j = 0; j < 8; ++j) {
      int qr = qr0 + j + ((lane < 16) ? 0 : 8);
      O[((size_t)(b * AS + qr) * AH + h) * AHD + dt * 16 + r16] =
          Oacc[dt][j] * l[j];
    }
  }
}

// ---------------------------------------------------------------------------
// Host-side orchestration. Workspace layout (bytes, needs ~96 MB):
//   qf  fp32 [4096,2048]  @ 0          (reused as attention output O)
//   kf  fp32 [4096,1024]  @ 33554432
//   vf  fp32 [4096,1024]  @ 50331648
//   qh  f16  [4096,2048]  @ 67108864   (RoPE'd, pre-scaled by 1/16)
//   kh  f16  [4096,1024]  @ 83886080   (RoPE'd)
//   vh  f16  [4096,1024]  @ 92274688
// ---------------------------------------------------------------------------
extern "C" void kernel_launch(void* const* d_in, const int* in_sizes, int n_in,
                              void* d_out, int out_size, void* d_ws, size_t ws_size,
                              hipStream_t stream) {
  const float* X     = (const float*)d_in[0];
  const int*   amask = (const int*)  d_in[1];
  const int*   pos   = (const int*)  d_in[2];
  const float* Wq    = (const float*)d_in[3];
  const float* Wk    = (const float*)d_in[4];
  const float* Wv    = (const float*)d_in[5];
  const float* Wo    = (const float*)d_in[6];
  (void)in_sizes; (void)n_in; (void)out_size; (void)ws_size;

  char* ws = (char*)d_ws;
  float*    qf = (float*)(ws);
  float*    kf = (float*)(ws + 33554432);
  float*    vf = (float*)(ws + 50331648);
  _Float16* qh = (_Float16*)(ws + 67108864);
  _Float16* kh = (_Float16*)(ws + 83886080);
  _Float16* vh = (_Float16*)(ws + 92274688);
  float*    O  = qf;                       // fp32 Q dead after RoPE/convert

  const int M = 4096;                      // B*S
  dim3 blk(256);

  // QKV projections
  gemm_hf_wmma<<<dim3(2048 / 128, M / 64), blk, 0, stream>>>(X, Wq, qf, M, 2048, 2304);
  gemm_hf_wmma<<<dim3(1024 / 128, M / 64), blk, 0, stream>>>(X, Wk, kf, M, 1024, 2304);
  gemm_hf_wmma<<<dim3(1024 / 128, M / 64), blk, 0, stream>>>(X, Wv, vf, M, 1024, 2304);

  // RoPE + f16 convert (q also scaled by 1/sqrt(256) = 0.0625)
  rope_f16_kernel<<<4194304 / 256, 256, 0, stream>>>(qf, pos, qh, 8, 0.0625f, 4194304);
  rope_f16_kernel<<<2097152 / 256, 256, 0, stream>>>(kf, pos, kh, 4, 1.0f,    2097152);
  cvt_f16_kernel <<<4194304 / 256, 256, 0, stream>>>(vf, vh, 4194304);

  // causal GQA attention with softcap (TDM-staged K tiles)
  flash_attn_kernel<<<dim3(2048 / 64, 8, 2), 128, 0, stream>>>(qh, kh, vh, amask, O);

  // output projection
  gemm_hf_wmma<<<dim3(2304 / 128, M / 64), blk, 0, stream>>>(O, Wo, (float*)d_out, M, 2304, 2048);
}
